// FullSelfAttention_36352603193508
// MI455X (gfx1250) — matrix operand order repaired and checked by reading, not compiled, not run on hardware
//
#include <hip/hip_runtime.h>

// MI455X / gfx1250, wave32. Fused self-attention:
//   cvt : one-shot f32->bf16 of x and Wq/Wk/Wv (hoists all conversions off hot path)
//   proj: q,k (row-major bf16) + vT (transposed bf16) via v_wmma_f32_16x16x32_bf16
//   attn: flash-attention per 16-query tile, 32-key blocks, fused streaming top-32
// Outputs (concat): idx int32 [4,4096,32] then out f32 [4,128,4096].
// Workspace: xbf 4MB + W 96KB + q/k/vT 12MB ~= 16 MB.

#define BATCH 4
#define SEQ   4096
#define DIM   128
#define TOPK  32

typedef __bf16 bf16_t;
typedef __attribute__((ext_vector_type(8)))  float  v8f;
typedef __attribute__((ext_vector_type(8)))  bf16_t v8bf;
typedef __attribute__((ext_vector_type(16))) bf16_t v16bf;

__device__ __forceinline__ bf16_t f2bf(float f) {
  return static_cast<bf16_t>(f);   // native fptrunc -> v_cvt on gfx1250
}

__device__ __forceinline__ v16bf cat8(v8bf a, v8bf b) {
  v16bf r;
#pragma unroll
  for (int i = 0; i < 8; ++i) { r[i] = a[i]; r[i + 8] = b[i]; }
  return r;
}

__device__ __forceinline__ v8f wmma_bf16(v16bf a, v16bf b, v8f c) {
  // D = A(16x32 bf16) * B(32x16 bf16) + C(16x16 f32)
  return __builtin_amdgcn_wmma_f32_16x16x32_bf16(
      /*neg_a=*/false, a, /*neg_b=*/false, b,
      /*c_mod=*/(short)0, c, /*reuse_a=*/false, /*reuse_b=*/false);
}

// xor-swizzle within the wave32 (ds_swizzle group-of-32: offset = xor<<10 | and 0x1f)
template <int XORM>
__device__ __forceinline__ float swzf(float v) {
  int i = __builtin_bit_cast(int, v);
  i = __builtin_amdgcn_ds_swizzle(i, (XORM << 10) | 0x1f);
  return __builtin_bit_cast(float, i);
}
// reductions stay inside each 16-lane half (xor masks < 16)
__device__ __forceinline__ float redmax16(float v) {
  v = fmaxf(v, swzf<1>(v)); v = fmaxf(v, swzf<2>(v));
  v = fmaxf(v, swzf<4>(v)); v = fmaxf(v, swzf<8>(v));
  return v;
}
__device__ __forceinline__ float redsum16(float v) {
  v += swzf<1>(v); v += swzf<2>(v); v += swzf<4>(v); v += swzf<8>(v);
  return v;
}

// ---------------------------------------------------------------------------
// Kernel 0: one-shot f32 -> bf16 conversion (8 elements per thread, B128 out)
// ---------------------------------------------------------------------------
__global__ __launch_bounds__(256) void cvt_kernel(const float* __restrict__ src,
                                                  bf16_t* __restrict__ dst, int n) {
  const int i = (blockIdx.x * 256 + threadIdx.x) * 8;
  if (i < n) {
    v8f a = *(const v8f*)(src + i);
    v8bf o;
#pragma unroll
    for (int j = 0; j < 8; ++j) o[j] = f2bf(a[j]);
    *(v8bf*)(dst + i) = o;
  }
}

// ---------------------------------------------------------------------------
// Kernel 1: q = x Wq^T, k = x Wk^T (row-major bf16); vT = (x Wv^T)^T (bf16)
// All operands already bf16. One wave handles 16 rows of x.
// ---------------------------------------------------------------------------
__global__ __launch_bounds__(128) void qkv_kernel(
    const bf16_t* __restrict__ X,
    const bf16_t* __restrict__ Wq, const bf16_t* __restrict__ Wk,
    const bf16_t* __restrict__ Wv,
    bf16_t* __restrict__ Qo, bf16_t* __restrict__ Ko, bf16_t* __restrict__ VTo) {
  const int lane = threadIdx.x & 31;
  const int w    = threadIdx.x >> 5;
  const int half = lane >> 4;
  const int l16  = lane & 15;
  const int tile = blockIdx.x * 4 + w;
  const int b    = tile >> 8;           // 256 tiles per batch
  const int nb   = (tile & 255) << 4;   // first row of this tile

  // A-tiles of x (16x32 bf16 per 32-wide K chunk)
  v16bf ax[4];
  {
    const bf16_t* xr = X + ((size_t)(b * SEQ + nb + l16) * DIM);
#pragma unroll
    for (int c = 0; c < 4; ++c) {
      const int off0 = c * 32 + half * 16;  // A order used here (same as B): K{0-15}|{16-31}
      ax[c] = cat8(*(const v8bf*)(xr + off0), *(const v8bf*)(xr + off0 + 8));
    }
  }

  const bf16_t* Ws[3] = {Wq, Wk, Wv};
#pragma unroll
  for (int mtx = 0; mtx < 3; ++mtx) {
    const bf16_t* W = Ws[mtx];
#pragma unroll
    for (int et = 0; et < 8; ++et) {
      const int e = et * 16 + l16;  // B layout: lane = output column
      v8f acc = {};
#pragma unroll
      for (int c = 0; c < 4; ++c) {
        // B[k=d][n=e] = W[e][d]; per half K = c*32 + half*16 + 0..15 contiguous
        const bf16_t* wp = W + (size_t)e * DIM + c * 32 + half * 16;
        v16bf bw = cat8(*(const v8bf*)wp, *(const v8bf*)(wp + 8));
        acc = wmma_bf16(ax[c], bw, acc);
      }
#pragma unroll
      for (int r = 0; r < 8; ++r) {      // C layout: VGPR r -> row r + half*8
        const int nrow = nb + r + half * 8;
        bf16_t val = f2bf(acc[r]);
        if (mtx == 0)      Qo[((size_t)(b * SEQ + nrow)) * DIM + e] = val;
        else if (mtx == 1) Ko[((size_t)(b * SEQ + nrow)) * DIM + e] = val;
        else               VTo[((size_t)(b * DIM + e)) * SEQ + nrow] = val;
      }
    }
  }
}

// ---------------------------------------------------------------------------
// Kernel 2: flash attention + fused streaming top-32 (monotonicity of softmax
// means top-k of attn == top-k of scaled scores). 1 wave per 16-query tile.
// ---------------------------------------------------------------------------
__global__ __launch_bounds__(128) void attn_kernel(
    const bf16_t* __restrict__ Q, const bf16_t* __restrict__ Km,
    const bf16_t* __restrict__ Vt,
    int* __restrict__ idxOut, float* __restrict__ outF) {
  __shared__ float  sS[4][16][33];   // raw scaled scores (padded rows)
  __shared__ bf16_t sP[4][16][32];   // softmax probabilities for PV A-tile
  __shared__ float  tkV[4][16][TOPK];
  __shared__ int    tkI[4][16][TOPK];

  const int lane = threadIdx.x & 31;
  const int w    = threadIdx.x >> 5;
  const int half = lane >> 4;
  const int l16  = lane & 15;
  const int tile = blockIdx.x * 4 + w;
  const int b    = tile >> 8;
  const int qb   = (tile & 255) << 4;

  // Q A-tiles held in registers for the whole K sweep (4 x v16bf = 32 VGPRs)
  v16bf aq[4];
  {
    const bf16_t* qrow = Q + ((size_t)(b * SEQ + qb + l16) * DIM);
#pragma unroll
    for (int c = 0; c < 4; ++c) {
      const int off0 = c * 32 + half * 16;
      aq[c] = cat8(*(const v8bf*)(qrow + off0), *(const v8bf*)(qrow + off0 + 8));
    }
  }

  v8f o[8] = {};                // 16x128 f32 accumulator (8 C-tiles over d)
  float m[8], l[8];
#pragma unroll
  for (int r = 0; r < 8; ++r) { m[r] = -1e30f; l[r] = 0.0f; }

  if (lane < 16) {
#pragma unroll
    for (int j = 0; j < TOPK; ++j) { tkV[w][lane][j] = -1e30f; tkI[w][lane][j] = 0; }
  }
  float minv = -1e30f; int minpos = 0;  // per-row list minimum (lanes 0..15)

  const float RS  = 0.08838834764831845f;  // 1/sqrt(128)
  const float L2E = 1.4426950408889634f;

#pragma unroll 1
  for (int kb = 0; kb < SEQ / 32; ++kb) {
    const int kbase = kb * 32;

    // ---- scores: S(16x32) = Q(16x128) * K^T, two 16-key sub-tiles -------
    v8f s0 = {}, s1 = {};
#pragma unroll
    for (int c = 0; c < 4; ++c) {
      const int d0 = c * 32 + half * 16;  // B layout K range per half
      const bf16_t* kp0 = Km + ((size_t)(b * SEQ + kbase + l16) * DIM + d0);
      const bf16_t* kp1 = Km + ((size_t)(b * SEQ + kbase + 16 + l16) * DIM + d0);
      v16bf b0 = cat8(*(const v8bf*)kp0, *(const v8bf*)(kp0 + 8));
      v16bf b1 = cat8(*(const v8bf*)kp1, *(const v8bf*)(kp1 + 8));
      s0 = wmma_bf16(aq[c], b0, s0);
      s1 = wmma_bf16(aq[c], b1, s1);
    }
    if (kb + 1 < SEQ / 32) {  // global_prefetch_b8 for next K block
      __builtin_prefetch(Km + ((size_t)(b * SEQ + kbase + 32 + l16) * DIM), 0, 0);
      __builtin_prefetch(Vt + ((size_t)(b * DIM + l16) * SEQ + kbase + 32), 0, 0);
    }
#pragma unroll
    for (int r = 0; r < 8; ++r) { s0[r] *= RS; s1[r] *= RS; }

    // ---- stage raw scores for top-k (C layout -> row-major in LDS) ------
#pragma unroll
    for (int r = 0; r < 8; ++r) {
      sS[w][r + half * 8][l16]      = s0[r];
      sS[w][r + half * 8][16 + l16] = s1[r];
    }
    asm volatile("s_wait_dscnt 0x0" ::: "memory");  // wave-local LDS handoff

    // ---- streaming top-32: lanes 0..15 each own one query row ----------
    if (lane < 16) {
      const int row = lane;
#pragma unroll 1
      for (int cc = 0; cc < 32; ++cc) {
        float sv = sS[w][row][cc];
        if (sv > minv) {
          tkV[w][row][minpos] = sv;
          tkI[w][row][minpos] = kbase + cc;
          float mv = tkV[w][row][0]; int mp = 0;
#pragma unroll 1
          for (int j = 1; j < TOPK; ++j) {
            float t = tkV[w][row][j];
            if (t < mv) { mv = t; mp = j; }
          }
          minv = mv; minpos = mp;
        }
      }
    }

    // ---- online softmax update -----------------------------------------
#pragma unroll
    for (int r = 0; r < 8; ++r) {
      float rm   = redmax16(fmaxf(s0[r], s1[r]));
      float mn   = fmaxf(m[r], rm);
      float corr = exp2f((m[r] - mn) * L2E);
      float p0   = exp2f((s0[r] - mn) * L2E);
      float p1   = exp2f((s1[r] - mn) * L2E);
      l[r] = l[r] * corr + redsum16(p0 + p1);
      m[r] = mn;
      s0[r] = p0; s1[r] = p1;
#pragma unroll
      for (int t = 0; t < 8; ++t) o[t][r] *= corr;
    }

    // ---- P: C layout -> A layout via LDS (bf16) -------------------------
#pragma unroll
    for (int r = 0; r < 8; ++r) {
      sP[w][r + half * 8][l16]      = f2bf(s0[r]);
      sP[w][r + half * 8][16 + l16] = f2bf(s1[r]);
    }
    asm volatile("s_wait_dscnt 0x0" ::: "memory");

    v16bf ap;
#pragma unroll
    for (int e = 0; e < 16; ++e) {
      const int j  = e >> 1;
      const int kk = half * 16 + e;
      ap[e] = sP[w][l16][kk];
    }

    // ---- O += P(16x32) * V(32x16 per d-tile), B from pre-transposed vT --
#pragma unroll
    for (int t = 0; t < 8; ++t) {
      const bf16_t* vp =
          Vt + ((size_t)(b * DIM + t * 16 + l16) * SEQ + kbase + half * 16);
      v16bf bv = cat8(*(const v8bf*)vp, *(const v8bf*)(vp + 8));
      o[t] = wmma_bf16(ap, bv, o[t]);
    }
  }

  // ---- epilogue: normalize, store transposed out[b][d][n] ---------------
#pragma unroll
  for (int r = 0; r < 8; ++r) {
    const float inv = 1.0f / l[r];
    const int nrow = qb + r + half * 8;
#pragma unroll
    for (int t = 0; t < 8; ++t) {
      outF[((size_t)(b * DIM + t * 16 + l16)) * SEQ + nrow] = o[t][r] * inv;
    }
  }

  // ---- top-k: sort 32 entries descending, emit indices -------------------
  if (lane < 16) {
    const int row = lane;
#pragma unroll 1
    for (int a2 = 0; a2 < TOPK; ++a2) {
      float bv = tkV[w][row][a2]; int bi = tkI[w][row][a2]; int bp = a2;
#pragma unroll 1
      for (int j = a2 + 1; j < TOPK; ++j) {
        float t = tkV[w][row][j];
        if (t > bv) { bv = t; bi = tkI[w][row][j]; bp = j; }
      }
      tkV[w][row][bp] = tkV[w][row][a2]; tkI[w][row][bp] = tkI[w][row][a2];
      tkV[w][row][a2] = bv;              tkI[w][row][a2] = bi;
      idxOut[((size_t)(b * SEQ + qb + row)) * TOPK + a2] = bi;
    }
  }
}

extern "C" void kernel_launch(void* const* d_in, const int* in_sizes, int n_in,
                              void* d_out, int out_size, void* d_ws, size_t ws_size,
                              hipStream_t stream) {
  const float* x  = (const float*)d_in[0];
  const float* wq = (const float*)d_in[1];
  const float* wk = (const float*)d_in[2];
  const float* wv = (const float*)d_in[3];

  const size_t NX = (size_t)BATCH * SEQ * DIM;   // 2,097,152
  const size_t NW = (size_t)DIM * DIM;           // 16,384

  bf16_t* q    = (bf16_t*)d_ws;
  bf16_t* k    = q   + NX;
  bf16_t* vT   = k   + NX;
  bf16_t* xbf  = vT  + NX;
  bf16_t* wqbf = xbf + NX;
  bf16_t* wkbf = wqbf + NW;
  bf16_t* wvbf = wkbf + NW;

  int*   idxOut = (int*)d_out;                                // [4,4096,32] int32
  float* outF   = (float*)d_out + (size_t)BATCH * SEQ * TOPK; // [4,128,4096] f32

  // one-shot bf16 conversion (x: 1024 blocks, W: 8 blocks each)
  cvt_kernel<<<(int)(NX / 8 / 256), 256, 0, stream>>>(x, xbf, (int)NX);
  cvt_kernel<<<(int)(NW / 8 / 256), 256, 0, stream>>>(wq, wqbf, (int)NW);
  cvt_kernel<<<(int)(NW / 8 / 256), 256, 0, stream>>>(wk, wkbf, (int)NW);
  cvt_kernel<<<(int)(NW / 8 / 256), 256, 0, stream>>>(wv, wvbf, (int)NW);

  // 1024 16-row tiles, 4 waves (128 threads) per block
  qkv_kernel<<<256, 128, 0, stream>>>(xbf, wqbf, wkbf, wvbf, q, k, vT);
  attn_kernel<<<256, 128, 0, stream>>>(q, k, vT, idxOut, outF);
}
